// EdgeBiasedMultiHeadAttention_2241972928603
// MI455X (gfx1250) — matrix operand order repaired and checked by reading, not compiled, not run on hardware
//
#include <hip/hip_runtime.h>
#include <math.h>

// ---------------------------------------------------------------------------
// EdgeBiasedMultiHeadAttention forward for MI455X (gfx1250, wave32, WMMA)
//   B=4, N=1024, D=1024, H=16, d=64
// Pipeline: cvt(x) / cvtT(W_in) / cvtT(W_out)
//   -> QKV WMMA GEMM   (block-cooperative, async-to-LDS staged B (K=64 tiles),
//                       double buffered, 8 WMMAs per barrier period, all
//                       fragments hoisted so DS loads clause up)
//   -> flash attention (block shares (b,h): async-staged K/V tiles, WMMA,
//                       streaming softmax, fp32 edge-bias stream)
//   -> output WMMA GEMM (same cooperative scheme), fp32 out
// ---------------------------------------------------------------------------

typedef __attribute__((ext_vector_type(16))) _Float16 v16h;
typedef __attribute__((ext_vector_type(8)))  _Float16 v8h;
typedef __attribute__((ext_vector_type(4)))  _Float16 v4h;
typedef __attribute__((ext_vector_type(8)))  float    v8f;
typedef __attribute__((ext_vector_type(4)))  float    v4f;

#define WMMA_F16(a, b, c) \
  __builtin_amdgcn_wmma_f32_16x16x32_f16(false, (a), false, (b), (short)0, (c), false, false)

static __device__ __forceinline__ v16h cat16(v8h lo, v8h hi) {
  return __builtin_shufflevector(lo, hi, 0,1,2,3,4,5,6,7,8,9,10,11,12,13,14,15);
}

// 16-byte async copy global -> LDS (per lane). ASYNCcnt-tracked; the VDST
// operand is a VGPR holding the LDS byte address (low 32 bits of generic ptr).
static __device__ __forceinline__ void async_cp16(void* lds, const void* gmem) {
  unsigned int       l = (unsigned int)(unsigned long long)(uintptr_t)lds;
  unsigned long long g = (unsigned long long)(uintptr_t)gmem;
  asm volatile("global_load_async_to_lds_b128 %0, %1, off" :: "v"(l), "v"(g) : "memory");
}
#define WAIT_ASYNC(n) asm volatile("s_wait_asynccnt " #n ::: "memory")

// ---------------------------------------------------------------------------
// f32 -> f16 elementwise (vectorized x4)
// ---------------------------------------------------------------------------
__global__ void k_cvt_f16(const float* __restrict__ src, _Float16* __restrict__ dst, int n4) {
  int i = blockIdx.x * blockDim.x + threadIdx.x;
  if (i < n4) {
    v4f f = ((const v4f*)src)[i];
    v4h h;
    h.x = (_Float16)f.x; h.y = (_Float16)f.y; h.z = (_Float16)f.z; h.w = (_Float16)f.w;
    ((v4h*)dst)[i] = h;
  }
}

// ---------------------------------------------------------------------------
// Wt[n*K + k] = (f16) W[k*NN + n]   (transpose+convert so B-tiles are contiguous)
// ---------------------------------------------------------------------------
__global__ void k_transpose_cvt(const float* __restrict__ W, _Float16* __restrict__ Wt,
                                int K, int NN) {
  long i = (long)blockIdx.x * blockDim.x + threadIdx.x;
  long total = (long)K * NN;
  if (i < total) {
    int n = (int)(i / K);
    int k = (int)(i % K);
    Wt[i] = (_Float16)W[(size_t)k * NN + n];
  }
}

// ---------------------------------------------------------------------------
// Shared GEMM mainloop body: block = 128(M) x 64(N), B tile 64(N) x 64(K)
// staged in LDS by async copies, double buffered. All A/B fragments of a
// K=64 step are hoisted (live simultaneously) so the 16 DS b128 loads clause
// together and the 8 WMMAs issue back-to-back.
// ---------------------------------------------------------------------------
static __device__ __forceinline__ void gemm_mainloop_128x64(
    const _Float16* __restrict__ Ar,    // A row base for this wave (+row*K)
    const _Float16* __restrict__ gB,    // staging base for this thread
    _Float16* __restrict__ bst0, _Float16* __restrict__ bst1,
    int K, int row, int kbA, int kbB, int c0r, int c0c, int c1r, int c1c,
    v8f acc[4]) {
  // prime buffer 0 with k-range [0,64)
  async_cp16(bst0 + c0r * 64 + c0c, gB + (size_t)c0r * K + c0c);
  async_cp16(bst0 + c1r * 64 + c1c, gB + (size_t)c1r * K + c1c);

  for (int k0 = 0; k0 < K; k0 += 64) {
    _Float16* bcur = ((k0 >> 6) & 1) ? bst1 : bst0;
    if (k0 + 64 < K) {
      _Float16* bnxt = ((k0 >> 6) & 1) ? bst0 : bst1;
      async_cp16(bnxt + c0r * 64 + c0c, gB + (size_t)c0r * K + k0 + 64 + c0c);
      async_cp16(bnxt + c1r * 64 + c1c, gB + (size_t)c1r * K + k0 + 64 + c1c);
      WAIT_ASYNC(0x2);                       // current tile landed
    } else {
      WAIT_ASYNC(0x0);
    }
    __syncthreads();

    // hoist both A fragments (k0..k0+31, k0+32..k0+63)
    v16h a0 = *(const v16h*)(Ar + k0 + kbA);
    v16h a1 = *(const v16h*)(Ar + k0 + 32 + kbA);
    // hoist all 8 B fragments -> distinct registers, clause-able DS loads
    v16h b00 = *(const v16h*)(bcur + (0 * 16 + row) * 64 + kbB);
    v16h b01 = *(const v16h*)(bcur + (1 * 16 + row) * 64 + kbB);
    v16h b02 = *(const v16h*)(bcur + (2 * 16 + row) * 64 + kbB);
    v16h b03 = *(const v16h*)(bcur + (3 * 16 + row) * 64 + kbB);
    v16h b10 = *(const v16h*)(bcur + (0 * 16 + row) * 64 + 32 + kbB);
    v16h b11 = *(const v16h*)(bcur + (1 * 16 + row) * 64 + 32 + kbB);
    v16h b12 = *(const v16h*)(bcur + (2 * 16 + row) * 64 + 32 + kbB);
    v16h b13 = *(const v16h*)(bcur + (3 * 16 + row) * 64 + 32 + kbB);

    acc[0] = WMMA_F16(a0, b00, acc[0]);
    acc[1] = WMMA_F16(a0, b01, acc[1]);
    acc[2] = WMMA_F16(a0, b02, acc[2]);
    acc[3] = WMMA_F16(a0, b03, acc[3]);
    acc[0] = WMMA_F16(a1, b10, acc[0]);
    acc[1] = WMMA_F16(a1, b11, acc[1]);
    acc[2] = WMMA_F16(a1, b12, acc[2]);
    acc[3] = WMMA_F16(a1, b13, acc[3]);

    __syncthreads();                         // all reads done before overwrite
  }
}

// ---------------------------------------------------------------------------
// QKV GEMM: C[4096,3072] = Xf16[4096,1024] @ W_in + b_in, scattered into
//   q -> [B,H,N,64] f16, k -> [B,H,N,64] f16, v^T -> [B,H,64,N] f16
// Grid: 32 m-blocks x 48 n-blocks = 1536 blocks x 256 threads.
// ---------------------------------------------------------------------------
__global__ __launch_bounds__(256) void k_qkv_gemm(
    const _Float16* __restrict__ A,   // [4096,1024]
    const _Float16* __restrict__ Bt,  // [3072,1024]  (= W_in^T)
    const float*    __restrict__ bias,// [3072]
    _Float16* __restrict__ qf, _Float16* __restrict__ kf, _Float16* __restrict__ vt) {
  const int K = 1024;
  __shared__ __align__(32) _Float16 bst[2][64 * 64];   // 2 x 8KB B tiles

  const int tid  = threadIdx.x;
  const int lane = tid & 31;
  const int lwid = tid >> 5;
  const int nb = blockIdx.x % 48, mb = blockIdx.x / 48;
  const int n0 = nb * 64;
  const int m0 = mb * 128 + lwid * 16;
  const int row = lane & 15, hi = lane >> 4;
  const int kbA = hi * 16, kbB = hi * 16;

  // staging: 64 rows x 8 chunks of 16B = 512 chunks, 2 per thread
  const int c0 = tid, c1 = tid + 256;
  const int c0r = c0 >> 3, c0c = (c0 & 7) * 8;
  const int c1r = c1 >> 3, c1c = (c1 & 7) * 8;
  const _Float16* gB = Bt + (size_t)n0 * K;

  v8f acc[4] = {};
  gemm_mainloop_128x64(A + (size_t)(m0 + row) * K, gB, bst[0], bst[1],
                       K, row, kbA, kbB, c0r, c0c, c1r, c1c, acc);

#pragma unroll
  for (int t = 0; t < 4; ++t) {
    int ng = n0 + t * 16 + row;          // output column of this lane
    int sel = ng >> 10;                  // 0=q 1=k 2=v
    int nc  = ng & 1023;
    int h = nc >> 6, dimi = nc & 63;
    float bv = bias[ng];
#pragma unroll
    for (int r = 0; r < 8; ++r) {
      int m = m0 + r + 8 * hi;
      int b = m >> 10, tok = m & 1023;
      _Float16 hv = (_Float16)(acc[t][r] + bv);
      size_t bh = (size_t)b * 16 + h;
      if (sel == 0)      qf[(bh * 1024 + tok) * 64 + dimi] = hv;
      else if (sel == 1) kf[(bh * 1024 + tok) * 64 + dimi] = hv;
      else               vt[(bh * 64 + dimi) * 1024 + tok] = hv;
    }
  }
}

// ---------------------------------------------------------------------------
// Flash attention with fp32 edge bias. Block = one (b,h) with 8 consecutive
// 16-query tiles (one per wave); K/V tiles async-staged to LDS and shared by
// all 8 waves, double buffered. 512 blocks x 256 threads. 256 WMMAs/wave.
// ---------------------------------------------------------------------------
__global__ __launch_bounds__(256) void k_attn(
    const float* __restrict__ ebias,          // [4,16,1024,1024]
    const unsigned char* __restrict__ kmask,  // [4,1024] bool
    const _Float16* __restrict__ qf,
    const _Float16* __restrict__ kf,
    const _Float16* __restrict__ vtp,         // [B,H,64,N]
    _Float16* __restrict__ of) {              // [B,N,1024] f16 (head concat)
  const int Nn = 1024, Dh = 64;
  __shared__ __align__(32) _Float16 kst[2][32 * 64];   // K tile: 32 keys x 64 d
  __shared__ __align__(32) _Float16 vst[2][64 * 32];   // V^T tile: 64 d x 32 keys
  __shared__ __align__(32) _Float16 pbuf[8 * 16 * 32]; // per-wave P scratch

  const int tid  = threadIdx.x;
  const int lane = tid & 31;
  const int lwid = tid >> 5;
  const int bh = blockIdx.x >> 3;             // 0..63 (b*16+h)
  const int qg = blockIdx.x & 7;
  const int h = bh & 15, b = bh >> 4;
  const int q0 = (qg * 8 + lwid) * 16;
  const int row = lane & 15, hi = lane >> 4;
  const int kbA = hi * 16, kbB = hi * 16;
  _Float16* pb = &pbuf[lwid * 512];

  // Q tile (16x64): two A-fragments for d=0..31 and d=32..63
  const _Float16* qp = qf + ((size_t)bh * Nn + q0) * Dh + (size_t)row * Dh;
  v16h qa0 = *(const v16h*)(qp + kbA);
  v16h qa1 = *(const v16h*)(qp + 32 + kbA);

  const _Float16* kp = kf  + (size_t)bh * Nn * Dh;
  const _Float16* vp = vtp + (size_t)bh * Dh * Nn;
  const float*    bp = ebias + ((size_t)bh * Nn + q0) * Nn;
  const unsigned char* mp = kmask + (size_t)b * Nn;

  // staging coords: K tile 32 rows x 8 chunks; V tile 64 rows x 4 chunks
  const int krow = tid >> 3, kcol = (tid & 7) * 8;
  const int vrow = tid >> 2, vcol = (tid & 3) * 8;

  async_cp16(&kst[0][krow * 64 + kcol], kp + (size_t)krow * Dh + kcol);
  async_cp16(&vst[0][vrow * 32 + vcol], vp + (size_t)vrow * Nn + vcol);

  v8f o[4] = {};
  float m_i[8], l_i[8];
#pragma unroll
  for (int r = 0; r < 8; ++r) { m_i[r] = -__builtin_inff(); l_i[r] = 0.f; }

  for (int kb = 0; kb < Nn; kb += 32) {
    const int cur = (kb >> 5) & 1;
    if (kb + 32 < Nn) {
      const int nxt = cur ^ 1;
      async_cp16(&kst[nxt][krow * 64 + kcol], kp + (size_t)(kb + 32 + krow) * Dh + kcol);
      async_cp16(&vst[nxt][vrow * 32 + vcol], vp + (size_t)vrow * Nn + kb + 32 + vcol);
      WAIT_ASYNC(0x2);                       // current pair landed
    } else {
      WAIT_ASYNC(0x0);
    }
    __syncthreads();

    const _Float16* kc = kst[cur];
    const _Float16* vc = vst[cur];
    const int key0 = kb + row;
    const int key1 = key0 + 16;

    // K-tile fragments (distinct registers -> clause-able DS loads)
    v16h kb00 = *(const v16h*)(kc + row * 64 + kbB);
    v16h kb01 = *(const v16h*)(kc + row * 64 + 32 + kbB);
    v16h kb10 = *(const v16h*)(kc + (row + 16) * 64 + kbB);
    v16h kb11 = *(const v16h*)(kc + (row + 16) * 64 + 32 + kbB);
    // V-tile fragments issued early; they retire under the softmax VALU work
    v16h vb0 = *(const v16h*)(vc + (0 * 16 + row) * 32 + kbB);
    v16h vb1 = *(const v16h*)(vc + (1 * 16 + row) * 32 + kbB);
    v16h vb2 = *(const v16h*)(vc + (2 * 16 + row) * 32 + kbB);
    v16h vb3 = *(const v16h*)(vc + (3 * 16 + row) * 32 + kbB);

    // S = Q K^T : interleave the two independent accumulate chains
    v8f z = {};
    v8f s0 = WMMA_F16(qa0, kb00, z);
    v8f s1 = WMMA_F16(qa0, kb10, z);
    s0 = WMMA_F16(qa1, kb01, s0);
    s1 = WMMA_F16(qa1, kb11, s1);

    // scale + edge bias (the 256MB fp32 stream) + key padding mask
    float madd0 = mp[key0] ? -__builtin_inff() : 0.f;
    float madd1 = mp[key1] ? -__builtin_inff() : 0.f;
    const float* bq = bp + (size_t)(hi * 8) * Nn;
    if (kb + 32 < Nn) __builtin_prefetch(bq + key0 + 32, 0, 1);  // global_prefetch
#pragma unroll
    for (int r = 0; r < 8; ++r) {
      s0[r] = s0[r] * 0.125f + bq[(size_t)r * Nn + key0] + madd0;
      s1[r] = s1[r] * 0.125f + bq[(size_t)r * Nn + key1] + madd1;
    }

    // streaming softmax: rows live across 16-lane halves -> width-16 xor reduce
    float mc[8];
#pragma unroll
    for (int r = 0; r < 8; ++r) mc[r] = fmaxf(s0[r], s1[r]);
#pragma unroll
    for (int xm = 1; xm < 16; xm <<= 1)
#pragma unroll
      for (int r = 0; r < 8; ++r) mc[r] = fmaxf(mc[r], __shfl_xor(mc[r], xm, 16));
    float alpha[8];
#pragma unroll
    for (int r = 0; r < 8; ++r) {
      float mn = fmaxf(m_i[r], mc[r]);
      alpha[r] = __expf(m_i[r] - mn);
      m_i[r] = mn;
    }
    float p0[8], p1[8], rs[8];
#pragma unroll
    for (int r = 0; r < 8; ++r) {
      p0[r] = __expf(s0[r] - m_i[r]);
      p1[r] = __expf(s1[r] - m_i[r]);
      rs[r] = p0[r] + p1[r];
    }
#pragma unroll
    for (int xm = 1; xm < 16; xm <<= 1)
#pragma unroll
      for (int r = 0; r < 8; ++r) rs[r] += __shfl_xor(rs[r], xm, 16);
#pragma unroll
    for (int r = 0; r < 8; ++r) l_i[r] = l_i[r] * alpha[r] + rs[r];
#pragma unroll
    for (int t = 0; t < 4; ++t)
#pragma unroll
      for (int r = 0; r < 8; ++r) o[t][r] *= alpha[r];

    // P: C-layout -> LDS -> A-layout (wave-private slab, DS ops in-order)
#pragma unroll
    for (int r = 0; r < 8; ++r) {
      int m = r + 8 * hi;
      pb[m * 32 + row]      = (_Float16)p0[r];
      pb[m * 32 + 16 + row] = (_Float16)p1[r];
    }
    asm volatile("s_wait_dscnt 0x0" ::: "memory");
    v16h pa = *(const v16h*)(pb + row * 32 + kbA);

    // O += P @ V : back-to-back WMMAs on independent accumulators
    o[0] = WMMA_F16(pa, vb0, o[0]);
    o[1] = WMMA_F16(pa, vb1, o[1]);
    o[2] = WMMA_F16(pa, vb2, o[2]);
    o[3] = WMMA_F16(pa, vb3, o[3]);
    __syncthreads();                          // reads done before buffer reuse
  }

  float rinv[8];
#pragma unroll
  for (int r = 0; r < 8; ++r) rinv[r] = 1.0f / l_i[r];
  _Float16* op = of + ((size_t)b * Nn + q0) * 1024 + (size_t)h * 64;
#pragma unroll
  for (int t = 0; t < 4; ++t)
#pragma unroll
    for (int r = 0; r < 8; ++r) {
      int m = r + 8 * hi;
      op[(size_t)m * 1024 + t * 16 + row] = (_Float16)(o[t][r] * rinv[r]);
    }
}

// ---------------------------------------------------------------------------
// Output projection: out[4096,1024] = attn_f16 @ W_out + b_out (fp32 out)
// Block = 128(M) x 64(N), async-staged B (K=64). 32 x 16 = 512 blocks.
// ---------------------------------------------------------------------------
__global__ __launch_bounds__(256) void k_out_gemm(
    const _Float16* __restrict__ A,   // [4096,1024]
    const _Float16* __restrict__ Bt,  // [1024,1024] (= W_out^T)
    const float*    __restrict__ bias,
    float* __restrict__ out) {
  const int K = 1024;
  __shared__ __align__(32) _Float16 bst[2][64 * 64];

  const int tid  = threadIdx.x;
  const int lane = tid & 31;
  const int lwid = tid >> 5;
  const int nb = blockIdx.x & 15, mb = blockIdx.x >> 4;
  const int n0 = nb * 64;
  const int m0 = mb * 128 + lwid * 16;
  const int row = lane & 15, hi = lane >> 4;
  const int kbA = hi * 16, kbB = hi * 16;

  const int c0 = tid, c1 = tid + 256;
  const int c0r = c0 >> 3, c0c = (c0 & 7) * 8;
  const int c1r = c1 >> 3, c1c = (c1 & 7) * 8;
  const _Float16* gB = Bt + (size_t)n0 * K;

  v8f acc[4] = {};
  gemm_mainloop_128x64(A + (size_t)(m0 + row) * K, gB, bst[0], bst[1],
                       K, row, kbA, kbB, c0r, c0c, c1r, c1c, acc);

#pragma unroll
  for (int t = 0; t < 4; ++t) {
    int ng = n0 + t * 16 + row;
    float bv = bias[ng];
#pragma unroll
    for (int r = 0; r < 8; ++r) {
      int m = m0 + r + 8 * hi;
      out[(size_t)m * 1024 + ng] = acc[t][r] + bv;
    }
  }
}

// ---------------------------------------------------------------------------
extern "C" void kernel_launch(void* const* d_in, const int* in_sizes, int n_in,
                              void* d_out, int out_size, void* d_ws, size_t ws_size,
                              hipStream_t stream) {
  const float* x     = (const float*)d_in[0];   // [4,1024,1024]
  const float* ebias = (const float*)d_in[1];   // [4,16,1024,1024]
  const unsigned char* kmask = (const unsigned char*)d_in[2];  // [4,1024] bool
  const float* W_in  = (const float*)d_in[3];   // [1024,3072]
  const float* b_in  = (const float*)d_in[4];   // [3072]
  const float* W_out = (const float*)d_in[5];   // [1024,1024]
  const float* b_out = (const float*)d_in[6];   // [1024]
  float* out = (float*)d_out;                   // [4,1024,1024]
  (void)in_sizes; (void)n_in; (void)out_size; (void)ws_size;

  char* ws = (char*)d_ws;
  size_t off = 0;
  auto alloc = [&](size_t bytes) -> char* {
    char* p = ws + off;
    off += (bytes + 255) & ~(size_t)255;
    return p;
  };
  _Float16* xf     = (_Float16*)alloc((size_t)4096 * 1024 * 2);       // x in f16
  _Float16* wt_in  = (_Float16*)alloc((size_t)3072 * 1024 * 2);       // W_in^T f16
  _Float16* wt_out = (_Float16*)alloc((size_t)1024 * 1024 * 2);       // W_out^T f16
  _Float16* qf     = (_Float16*)alloc((size_t)64 * 1024 * 64 * 2);    // [B,H,N,d]
  _Float16* kf     = (_Float16*)alloc((size_t)64 * 1024 * 64 * 2);    // [B,H,N,d]
  _Float16* vt     = (_Float16*)alloc((size_t)64 * 64 * 1024 * 2);    // [B,H,d,N]
  _Float16* af     = (_Float16*)alloc((size_t)4096 * 1024 * 2);       // attn out f16

  // 1) convert inputs
  k_cvt_f16<<<(4096 * 1024 / 4 + 255) / 256, 256, 0, stream>>>(x, xf, 4096 * 1024 / 4);
  k_transpose_cvt<<<(3072 * 1024 + 255) / 256, 256, 0, stream>>>(W_in, wt_in, 1024, 3072);
  k_transpose_cvt<<<(1024 * 1024 + 255) / 256, 256, 0, stream>>>(W_out, wt_out, 1024, 1024);
  // 2) QKV projection (WMMA, async-staged B)
  k_qkv_gemm<<<1536, 256, 0, stream>>>(xf, wt_in, b_in, qf, kf, vt);
  // 3) edge-biased flash attention (WMMA, async-staged K/V)
  k_attn<<<512, 256, 0, stream>>>(ebias, kmask, qf, kf, vt, af);
  // 4) output projection (WMMA, async-staged B)
  k_out_gemm<<<512, 256, 0, stream>>>(af, wt_out, b_out, out);
}
